// KuramotoPIGNN_V2_16535624090330
// MI455X (gfx1250) — matrix operand order repaired and checked by reading, not compiled, not run on hardware
//
#include <hip/hip_runtime.h>

// ---------------------------------------------------------------------------
// Kuramoto PIGNN forward for MI455X (gfx1250), bf16 WMMA path.
// All dense GEMMs run through v_wmma_f32_16x16x32_bf16 (wave32).
// B-fragment loads are software double-buffered so WMMAs overlap L2 latency
// (s_wait_loadcnt 0x1 instead of 0x0 serialization).
// ---------------------------------------------------------------------------

#define Hdim 128
#define THREADS 256

typedef __attribute__((ext_vector_type(16))) __bf16 v16bf;
typedef __attribute__((ext_vector_type(8)))  float  v8f;
typedef unsigned short u16;
typedef unsigned int   u32;

__device__ __forceinline__ u16 f2bf(float f) {
  u32 u = __float_as_uint(f);
  u += 0x7FFFu + ((u >> 16) & 1u);       // round-to-nearest-even
  return (u16)(u >> 16);
}
__device__ __forceinline__ float bf2f(u16 h) {
  return __uint_as_float(((u32)h) << 16);
}

union FragU { uint4 q[2]; v16bf v; };

// A-fragment (16x32 bf16): lanes 0-15 row M=lane hold K = k0+{0..7,16..23};
// lanes 16-31 row M=lane-16 hold K = k0+{8..15,24..31}.
__device__ __forceinline__ v16bf load_a_frag(const u16* act, int stride,
                                             int row_base, int k0, int lane) {
  const int m     = row_base + (lane & 15);
  const int khalf = (lane & 16) ? 8 : 0;
  const u16* p = act + m * stride + k0 + khalf;
  FragU f;
  f.q[0] = *(const uint4*)(p);        // K = k0+khalf+0..7
  f.q[1] = *(const uint4*)(p + 16);   // K = k0+khalf+16..23
  return f.v;
}

// B-fragment (32x16 bf16), pre-swizzled: lane l holds col=(l&15),
// K = kb*32 + ((l&16)?8:0) + {0..7,16..23}, contiguous 32 bytes per lane.
__device__ __forceinline__ v16bf load_b_frag(const u16* __restrict__ wfr,
                                             int kb, int n, int lane) {
  const u16* p = wfr + (((kb * 8 + n) * 32 + lane) << 4);
  FragU f;
  f.q[0] = *(const uint4*)(p);
  f.q[1] = *(const uint4*)(p + 8);
  return f.v;
}

// 16 (rows) x 128 (cols) output tile per wave; acc[n] = column tile n*16..n*16+15.
// B fragments double-buffered; next A fragment prefetched under the last WMMA
// of each k-block so the wave never stalls at loadcnt==0.
__device__ __forceinline__ void gemm_acc(const u16* act, int stride,
                                         const u16* __restrict__ wfr, int nkb,
                                         int lane, int row_base, v8f acc[8]) {
  v16bf a = load_a_frag(act, stride, row_base, 0, lane);
  v16bf b = load_b_frag(wfr, 0, 0, lane);
  for (int kb = 0; kb < nkb; ++kb) {
#pragma unroll
    for (int n = 0; n < 8; ++n) {
      v16bf b_nxt = b;
      v16bf a_nxt = a;
      if (n < 7) {
        b_nxt = load_b_frag(wfr, kb, n + 1, lane);
      } else if (kb + 1 < nkb) {
        b_nxt = load_b_frag(wfr, kb + 1, 0, lane);
        a_nxt = load_a_frag(act, stride, row_base, (kb + 1) * 32, lane);
      }
      acc[n] = __builtin_amdgcn_wmma_f32_16x16x32_bf16(
          false, a, false, b, (short)0, acc[n], false, false);
      b = b_nxt;
      a = a_nxt;
    }
  }
}

// C/D layout: vgpr i, lanes 0-15: row=i, col=lane; lanes 16-31: row=8+i, col=lane-16.
#define C_ROW(i, lane)  ((i) + (((lane) & 16) ? 8 : 0))
#define C_COL(n, lane)  ((n) * 16 + ((lane) & 15))

// ---------------------------------------------------------------------------
// Pre-swizzle a row-major fp32 weight W[K][128] into bf16 B-fragments.
// ---------------------------------------------------------------------------
__global__ void prep_weight_frags(const float* __restrict__ w,
                                  u16* __restrict__ dst, int K) {
  int tid = blockIdx.x * blockDim.x + threadIdx.x;
  if (tid >= K * Hdim) return;
  int j  = tid & 15;
  int l  = (tid >> 4) & 31;
  int n  = (tid >> 9) & 7;
  int kb = tid >> 12;
  int row = kb * 32 + ((l & 16) ? 8 : 0) + (j & 7) + ((j & 8) ? 16 : 0);
  int col = n * 16 + (l & 15);
  dst[tid] = f2bf(w[row * Hdim + col]);
}

// ---------------------------------------------------------------------------
// Node encoder: h = mlp3(x[:, :4]) ; h += pp(relu path)   -> h_bf16 [N,128]
// ---------------------------------------------------------------------------
__global__ void node_encode(const float* __restrict__ x,
                            const float* __restrict__ se_w1, const float* __restrict__ se_b1,
                            const u16* __restrict__ wf_se2, const float* __restrict__ se_b2,
                            const u16* __restrict__ wf_se3, const float* __restrict__ se_b3,
                            const u16* __restrict__ wf_pp1, const float* __restrict__ pp_b1,
                            const u16* __restrict__ wf_pp2, const float* __restrict__ pp_b2,
                            u16* __restrict__ h_bf16, int N) {
  extern __shared__ __align__(16) unsigned char smem[];
  u16*   bufA = (u16*)smem;             // 128 x 136 bf16
  u16*   bufB = bufA + 128 * 136;       // 128 x 136 bf16
  float* xs   = (float*)(bufB + 128 * 136);   // 128 x 4 staged inputs

  const int tid = threadIdx.x;
  const int lane = tid & 31;
  const int row_base = (tid >> 5) * 16;
  const int base = blockIdx.x * 128;

  // Stage x[:, :4] for this tile once.
  if (tid < 128) {
    int node = base + tid;
#pragma unroll
    for (int k = 0; k < 4; ++k)
      xs[tid * 4 + k] = (node < N) ? x[node * 10 + k] : 0.0f;
  }
  __syncthreads();

  // Layer 1: [128,4] @ [4,128], VALU (K too small for WMMA to pay off).
  for (int idx = tid; idx < 128 * 128; idx += THREADS) {
    int row = idx >> 7, col = idx & 127;
    float v = se_b1[col];
#pragma unroll
    for (int k = 0; k < 4; ++k)
      v = fmaf(xs[row * 4 + k], se_w1[k * Hdim + col], v);
    v = fmaxf(v, 0.0f);
    bufA[row * 136 + col] = f2bf(v);
  }
  __syncthreads();

  v8f acc[8], hacc[8];

  // Layer 2: relu(bufA @ se_w2 + b2) -> bufB
#pragma unroll
  for (int n = 0; n < 8; ++n) acc[n] = (v8f)0.0f;
  gemm_acc(bufA, 136, wf_se2, 4, lane, row_base, acc);
#pragma unroll
  for (int n = 0; n < 8; ++n) {
    int col = C_COL(n, lane);
    float b = se_b2[col];
#pragma unroll
    for (int i = 0; i < 8; ++i) {
      float v = fmaxf(acc[n][i] + b, 0.0f);
      bufB[(row_base + C_ROW(i, lane)) * 136 + col] = f2bf(v);
    }
  }

  // Layer 3: bufB @ se_w3 + b3 (no relu) -> keep in hacc, bf16 copy in bufA
#pragma unroll
  for (int n = 0; n < 8; ++n) acc[n] = (v8f)0.0f;
  gemm_acc(bufB, 136, wf_se3, 4, lane, row_base, acc);
#pragma unroll
  for (int n = 0; n < 8; ++n) {
    int col = C_COL(n, lane);
    float b = se_b3[col];
#pragma unroll
    for (int i = 0; i < 8; ++i) {
      float v = acc[n][i] + b;
      hacc[n][i] = v;
      bufA[(row_base + C_ROW(i, lane)) * 136 + col] = f2bf(v);
    }
  }

  // post_proj layer 1: relu(h @ pp_w1 + b1) -> bufB
#pragma unroll
  for (int n = 0; n < 8; ++n) acc[n] = (v8f)0.0f;
  gemm_acc(bufA, 136, wf_pp1, 4, lane, row_base, acc);
#pragma unroll
  for (int n = 0; n < 8; ++n) {
    int col = C_COL(n, lane);
    float b = pp_b1[col];
#pragma unroll
    for (int i = 0; i < 8; ++i) {
      float v = fmaxf(acc[n][i] + b, 0.0f);
      bufB[(row_base + C_ROW(i, lane)) * 136 + col] = f2bf(v);
    }
  }

  // post_proj layer 2 + residual: h = h + bufB @ pp_w2 + b2  -> h_bf16 global
#pragma unroll
  for (int n = 0; n < 8; ++n) acc[n] = (v8f)0.0f;
  gemm_acc(bufB, 136, wf_pp2, 4, lane, row_base, acc);
#pragma unroll
  for (int n = 0; n < 8; ++n) {
    int col = C_COL(n, lane);
    float b = pp_b2[col];
#pragma unroll
    for (int i = 0; i < 8; ++i) {
      int node = base + row_base + C_ROW(i, lane);
      if (node < N) {
        float v = hacc[n][i] + acc[n][i] + b;
        h_bf16[node * Hdim + col] = f2bf(v);
      }
    }
  }
}

// ---------------------------------------------------------------------------
// Edge MLP + masked segment-sum scatter into agg[N,128].
// ---------------------------------------------------------------------------
__global__ void edge_mp(const float* __restrict__ theta, const float* __restrict__ alive,
                        const float* __restrict__ Kp,
                        const int* __restrict__ ei, int E,
                        const u16* __restrict__ h_bf16,
                        const u16* __restrict__ wf_em1, const float* __restrict__ em_w1,
                        const float* __restrict__ em_b1,
                        const u16* __restrict__ wf_em2, const float* __restrict__ em_b2,
                        const u16* __restrict__ wf_em3, const float* __restrict__ em_b3,
                        float* __restrict__ agg) {
  extern __shared__ __align__(16) unsigned char smem[];
  u16*   bufA = (u16*)smem;                    // 128 x 264 bf16 (256 cols + pad)
  u16*   bufB = bufA + 128 * 264;              // 128 x 136 bf16
  float* sind = (float*)(bufB + 128 * 136);
  float* cosd = sind + 128;
  float* alv  = cosd + 128;
  int*   srce = (int*)(alv + 128);
  int*   dste = srce + 128;

  const int tid = threadIdx.x;
  const int lane = tid & 31;
  const int row_base = (tid >> 5) * 16;
  const int base = blockIdx.x * 128;

  if (tid < 128) {
    int e = base + tid;
    float sd = 0.f, cd = 0.f, a = 0.f;
    int s = 0, d = 0;
    if (e < E) {
      s = ei[e];
      d = ei[E + e];
      float del = theta[s] - theta[d];
      sd = sinf(del);
      cd = cosf(del);
      a = alive[s] * alive[d];
    }
    sind[tid] = sd; cosd[tid] = cd; alv[tid] = a;
    srce[tid] = s; dste[tid] = d;
  }
  __syncthreads();

  // Gather edge features: [ h[dst] | h[src] ] in bf16, 16B chunks, coalesced.
  for (int idx = tid; idx < 128 * 32; idx += THREADS) {
    int el = idx >> 5, c = idx & 31;
    uint4 q = make_uint4(0u, 0u, 0u, 0u);
    if (base + el < E) {
      int nidx = (c < 16) ? dste[el] : srce[el];
      q = *(const uint4*)(h_bf16 + nidx * Hdim + (c & 15) * 8);
    }
    *(uint4*)(bufA + el * 264 + c * 8) = q;
  }
  __syncthreads();

  v8f acc[8];
  const float Kv = Kp[0];

  // Edge layer 1: K = 256 via WMMA + rank-3 VALU tail (sin, cos, K columns).
#pragma unroll
  for (int n = 0; n < 8; ++n) acc[n] = (v8f)0.0f;
  gemm_acc(bufA, 264, wf_em1, 8, lane, row_base, acc);
#pragma unroll
  for (int n = 0; n < 8; ++n) {
    int col = C_COL(n, lane);
    float w256 = em_w1[256 * Hdim + col];
    float w257 = em_w1[257 * Hdim + col];
    float w258 = em_w1[258 * Hdim + col];
    float b = em_b1[col];
#pragma unroll
    for (int i = 0; i < 8; ++i) {
      int r = row_base + C_ROW(i, lane);
      float v = acc[n][i] + b;
      v = fmaf(sind[r], w256, v);
      v = fmaf(cosd[r], w257, v);
      v = fmaf(Kv, w258, v);
      bufB[r * 136 + col] = f2bf(fmaxf(v, 0.0f));
    }
  }

  // Edge layer 2
#pragma unroll
  for (int n = 0; n < 8; ++n) acc[n] = (v8f)0.0f;
  gemm_acc(bufB, 136, wf_em2, 4, lane, row_base, acc);
#pragma unroll
  for (int n = 0; n < 8; ++n) {
    int col = C_COL(n, lane);
    float b = em_b2[col];
#pragma unroll
    for (int i = 0; i < 8; ++i) {
      int r = row_base + C_ROW(i, lane);
      bufA[r * 264 + col] = f2bf(fmaxf(acc[n][i] + b, 0.0f));
    }
  }

  // Edge layer 3 + mask + atomic scatter (segment_sum over dst).
#pragma unroll
  for (int n = 0; n < 8; ++n) acc[n] = (v8f)0.0f;
  gemm_acc(bufA, 264, wf_em3, 4, lane, row_base, acc);
#pragma unroll
  for (int n = 0; n < 8; ++n) {
    int col = C_COL(n, lane);
    float b = em_b3[col];
#pragma unroll
    for (int i = 0; i < 8; ++i) {
      int r = row_base + C_ROW(i, lane);
      if (base + r < E) {
        float v = (acc[n][i] + b) * alv[r];
        atomicAdd(&agg[dste[r] * Hdim + col], v);
      }
    }
  }
}

// ---------------------------------------------------------------------------
// Node decoder: concat(h, agg, omega, sin, cos) -> MLP -> delta/theta_next.
// ---------------------------------------------------------------------------
__global__ void node_decode(const float* __restrict__ theta, const float* __restrict__ omega,
                            const float* __restrict__ alive,
                            const u16* __restrict__ h_bf16, const float* __restrict__ agg,
                            const u16* __restrict__ wf_nm1, const float* __restrict__ nm_w1,
                            const float* __restrict__ nm_b1,
                            const u16* __restrict__ wf_nm2, const float* __restrict__ nm_b2,
                            const float* __restrict__ nm_w3, const float* __restrict__ nm_b3,
                            float* __restrict__ out, int N) {
  extern __shared__ __align__(16) unsigned char smem[];
  u16*   bufA  = (u16*)smem;                   // 128 x 264 bf16
  u16*   bufB  = bufA + 128 * 264;             // 128 x 136 bf16
  float* th_s  = (float*)(bufB + 128 * 136);
  float* om_s  = th_s + 128;
  float* al_s  = om_s + 128;
  float* sth_s = al_s + 128;
  float* cth_s = sth_s + 128;

  const int tid = threadIdx.x;
  const int lane = tid & 31;
  const int row_base = (tid >> 5) * 16;
  const int base = blockIdx.x * 128;

  if (tid < 128) {
    int node = base + tid;
    float th = 0.f, om = 0.f, al = 0.f;
    if (node < N) { th = theta[node]; om = omega[node]; al = alive[node]; }
    th_s[tid] = th; om_s[tid] = om; al_s[tid] = al;
    sth_s[tid] = sinf(th); cth_s[tid] = cosf(th);
  }
  // Gather h (bf16) into cols 0..127
  for (int idx = tid; idx < 128 * 16; idx += THREADS) {
    int el = idx >> 4, c = idx & 15;
    int node = base + el;
    uint4 q = make_uint4(0u, 0u, 0u, 0u);
    if (node < N) q = *(const uint4*)(h_bf16 + node * Hdim + c * 8);
    *(uint4*)(bufA + el * 264 + c * 8) = q;
  }
  // agg (f32) -> bf16, cols 128..255
  for (int idx = tid; idx < 128 * 128; idx += THREADS) {
    int el = idx >> 7, c = idx & 127;
    int node = base + el;
    float v = (node < N) ? agg[node * Hdim + c] : 0.0f;
    bufA[el * 264 + 128 + c] = f2bf(v);
  }
  __syncthreads();

  v8f acc[8];

  // Decoder layer 1: K=256 WMMA + rank-3 tail (omega, sin, cos).
#pragma unroll
  for (int n = 0; n < 8; ++n) acc[n] = (v8f)0.0f;
  gemm_acc(bufA, 264, wf_nm1, 8, lane, row_base, acc);
#pragma unroll
  for (int n = 0; n < 8; ++n) {
    int col = C_COL(n, lane);
    float w256 = nm_w1[256 * Hdim + col];
    float w257 = nm_w1[257 * Hdim + col];
    float w258 = nm_w1[258 * Hdim + col];
    float b = nm_b1[col];
#pragma unroll
    for (int i = 0; i < 8; ++i) {
      int r = row_base + C_ROW(i, lane);
      float v = acc[n][i] + b;
      v = fmaf(om_s[r], w256, v);
      v = fmaf(sth_s[r], w257, v);
      v = fmaf(cth_s[r], w258, v);
      bufB[r * 136 + col] = f2bf(fmaxf(v, 0.0f));
    }
  }

  // Decoder layer 2
#pragma unroll
  for (int n = 0; n < 8; ++n) acc[n] = (v8f)0.0f;
  gemm_acc(bufB, 136, wf_nm2, 4, lane, row_base, acc);
#pragma unroll
  for (int n = 0; n < 8; ++n) {
    int col = C_COL(n, lane);
    float b = nm_b2[col];
#pragma unroll
    for (int i = 0; i < 8; ++i) {
      int r = row_base + C_ROW(i, lane);
      bufA[r * 264 + col] = f2bf(fmaxf(acc[n][i] + b, 0.0f));
    }
  }
  __syncthreads();

  // Final H->1 dot + mask + phase wrap.
  if (tid < 128) {
    int node = base + tid;
    if (node < N) {
      float s = nm_b3[0];
      for (int c = 0; c < Hdim; ++c)
        s = fmaf(bf2f(bufA[tid * 264 + c]), nm_w3[c], s);
      float delta = s * al_s[tid];
      out[node] = delta;
      float tn = th_s[tid] + delta;
      out[N + node] = atan2f(sinf(tn), cosf(tn));
    }
  }
}

// ---------------------------------------------------------------------------
// Host-side orchestration.
// ---------------------------------------------------------------------------
extern "C" void kernel_launch(void* const* d_in, const int* in_sizes, int n_in,
                              void* d_out, int out_size, void* d_ws, size_t ws_size,
                              hipStream_t stream) {
  const float* x     = (const float*)d_in[0];
  const float* theta = (const float*)d_in[1];
  const float* omega = (const float*)d_in[2];
  const float* alive = (const float*)d_in[3];
  const float* Kp    = (const float*)d_in[4];
  const int*   ei    = (const int*)d_in[5];
  const float* se_w1 = (const float*)d_in[6];  const float* se_b1 = (const float*)d_in[7];
  const float* se_w2 = (const float*)d_in[8];  const float* se_b2 = (const float*)d_in[9];
  const float* se_w3 = (const float*)d_in[10]; const float* se_b3 = (const float*)d_in[11];
  const float* pp_w1 = (const float*)d_in[12]; const float* pp_b1 = (const float*)d_in[13];
  const float* pp_w2 = (const float*)d_in[14]; const float* pp_b2 = (const float*)d_in[15];
  const float* em_w1 = (const float*)d_in[16]; const float* em_b1 = (const float*)d_in[17];
  const float* em_w2 = (const float*)d_in[18]; const float* em_b2 = (const float*)d_in[19];
  const float* em_w3 = (const float*)d_in[20]; const float* em_b3 = (const float*)d_in[21];
  const float* nm_w1 = (const float*)d_in[22]; const float* nm_b1 = (const float*)d_in[23];
  const float* nm_w2 = (const float*)d_in[24]; const float* nm_b2 = (const float*)d_in[25];
  const float* nm_w3 = (const float*)d_in[26]; const float* nm_b3 = (const float*)d_in[27];

  const int N = in_sizes[1];
  const int E = in_sizes[5] / 2;

  // Workspace carving (all 256B aligned): h_bf16, agg, 9 weight fragment blocks.
  unsigned char* ws = (unsigned char*)d_ws;
  size_t off = 0;
  auto carve = [&](size_t bytes) -> void* {
    void* p = ws + off;
    off += (bytes + 255) & ~(size_t)255;
    return p;
  };
  u16*   h_bf16 = (u16*)carve((size_t)N * Hdim * sizeof(u16));
  float* agg    = (float*)carve((size_t)N * Hdim * sizeof(float));
  u16* wf_se2 = (u16*)carve(128 * Hdim * sizeof(u16));
  u16* wf_se3 = (u16*)carve(128 * Hdim * sizeof(u16));
  u16* wf_pp1 = (u16*)carve(128 * Hdim * sizeof(u16));
  u16* wf_pp2 = (u16*)carve(128 * Hdim * sizeof(u16));
  u16* wf_em1 = (u16*)carve(256 * Hdim * sizeof(u16));
  u16* wf_em2 = (u16*)carve(128 * Hdim * sizeof(u16));
  u16* wf_em3 = (u16*)carve(128 * Hdim * sizeof(u16));
  u16* wf_nm1 = (u16*)carve(256 * Hdim * sizeof(u16));
  u16* wf_nm2 = (u16*)carve(128 * Hdim * sizeof(u16));
  (void)ws_size; (void)n_in; (void)out_size;

  auto prep = [&](const float* w, u16* dst, int K) {
    int total = K * Hdim;
    prep_weight_frags<<<(total + THREADS - 1) / THREADS, THREADS, 0, stream>>>(w, dst, K);
  };
  prep(se_w2, wf_se2, 128);
  prep(se_w3, wf_se3, 128);
  prep(pp_w1, wf_pp1, 128);
  prep(pp_w2, wf_pp2, 128);
  prep(em_w1, wf_em1, 256);   // first 256 rows; rows 256..258 handled as rank-3 tail
  prep(em_w2, wf_em2, 128);
  prep(em_w3, wf_em3, 128);
  prep(nm_w1, wf_nm1, 256);
  prep(nm_w2, wf_nm2, 128);

  hipMemsetAsync(agg, 0, (size_t)N * Hdim * sizeof(float), stream);

  const int ENC_SMEM = 2 * 128 * 136 * (int)sizeof(u16) + 128 * 4 * (int)sizeof(float); // 71680
  const int EDG_SMEM = 128 * 264 * 2 + 128 * 136 * 2 + 3 * 128 * 4 + 2 * 128 * 4;       // 104960
  const int DEC_SMEM = 128 * 264 * 2 + 128 * 136 * 2 + 5 * 128 * 4;                      // 104960

  node_encode<<<(N + 127) / 128, THREADS, ENC_SMEM, stream>>>(
      x, se_w1, se_b1, wf_se2, se_b2, wf_se3, se_b3,
      wf_pp1, pp_b1, wf_pp2, pp_b2, h_bf16, N);

  edge_mp<<<(E + 127) / 128, THREADS, EDG_SMEM, stream>>>(
      theta, alive, Kp, ei, E, h_bf16,
      wf_em1, em_w1, em_b1, wf_em2, em_b2, wf_em3, em_b3, agg);

  node_decode<<<(N + 127) / 128, THREADS, DEC_SMEM, stream>>>(
      theta, omega, alive, h_bf16, agg,
      wf_nm1, nm_w1, nm_b1, wf_nm2, nm_b2, nm_w3, nm_b3,
      (float*)d_out, N);
}
